// DHSMoEDetector_3092376453874
// MI455X (gfx1250) — matrix operand order repaired and checked by reading, not compiled, not run on hardware
//
#include <hip/hip_runtime.h>

#define N_TOK 16384
#define D_DIM 768
#define H_DIM 768
#define C_DIM 2
#define E_NUM 20
#define BM    64                 // tokens per workgroup tile
#define APAD  8                  // bf16 elems of row padding (16B) -> bank spread
#define ASTRIDE (D_DIM + APAD)   // 776 ushorts per LDS row
#define KCH   32                 // WMMA K per step (bf16)
#define NKCH  (D_DIM / KCH)      // 24
#define MAX_TILES 276            // floor(N/BM) + E upper bound
#define NTHREADS 512             // 16 wave32s

typedef __attribute__((ext_vector_type(16))) __bf16        v16bf;
typedef __attribute__((ext_vector_type(8)))  float         v8f;
typedef __attribute__((ext_vector_type(4)))  unsigned int  u32x4;
typedef __attribute__((ext_vector_type(4)))  float         f32x4;
typedef __attribute__((ext_vector_type(2)))  float         f32x2;

union frag_u { u32x4 q[2]; v16bf v; };

__device__ __forceinline__ unsigned int f2bf(float f) {
    unsigned int u = __float_as_uint(f);
    unsigned int r = u + 0x7FFFu + ((u >> 16) & 1u);   // round-to-nearest-even
    return r >> 16;
}

// ---- W1 [E][D][H] f32  ->  W1T [E][H][D] bf16 (tiled transpose) ----
__global__ __launch_bounds__(256) void k_w1_transpose(
    const float* __restrict__ W1, unsigned short* __restrict__ W1T)
{
    __shared__ float tile[32][33];
    int e  = blockIdx.z;
    int h0 = blockIdx.x * 32;
    int d0 = blockIdx.y * 32;
    const float* src = W1 + (size_t)e * D_DIM * H_DIM;
    #pragma unroll
    for (int i = threadIdx.y; i < 32; i += 8)
        tile[i][threadIdx.x] = src[(size_t)(d0 + i) * H_DIM + h0 + threadIdx.x];
    __syncthreads();
    unsigned short* dst = W1T + (size_t)e * H_DIM * D_DIM;
    #pragma unroll
    for (int i = threadIdx.y; i < 32; i += 8)
        dst[(size_t)(h0 + i) * D_DIM + d0 + threadIdx.x] =
            (unsigned short)f2bf(tile[threadIdx.x][i]);
}

// ---- token -> expert histogram ----
__global__ void k_hist(const int* __restrict__ idx, int* __restrict__ counts) {
    int i = blockIdx.x * blockDim.x + threadIdx.x;
    if (i < N_TOK) atomicAdd(&counts[idx[i]], 1);
}

// ---- scan + tile table (E=20, single thread is fine) ----
__global__ void k_scan(const int* __restrict__ counts, int* __restrict__ offsets,
                       int* __restrict__ numTiles, int* __restrict__ tileExpert,
                       int* __restrict__ tileStart)
{
    if (threadIdx.x == 0) {
        int off = 0;
        for (int e = 0; e < E_NUM; e++) { offsets[e] = off; off += counts[e]; }
        offsets[E_NUM] = off;
        int t = 0;
        for (int e = 0; e < E_NUM; e++)
            for (int s = offsets[e]; s < offsets[e + 1]; s += BM) {
                tileExpert[t] = e; tileStart[t] = s; t++;
            }
        numTiles[0] = t;
    }
}

// ---- scatter tokens into per-expert contiguous order ----
__global__ void k_scatter(const int* __restrict__ idx, const int* __restrict__ offsets,
                          int* __restrict__ cursors, int* __restrict__ perm)
{
    int i = blockIdx.x * blockDim.x + threadIdx.x;
    if (i < N_TOK) {
        int e = idx[i];
        int p = atomicAdd(&cursors[e], 1);
        perm[offsets[e] + p] = i;
    }
}

// ---- fused grouped-GEMM MoE: relu(x@W1+b1)@W2+b2, scatter to out ----
__global__ __launch_bounds__(NTHREADS) void k_moe_gemm(
    const float* __restrict__ emb, const unsigned short* __restrict__ W1T,
    const float* __restrict__ b1, const float* __restrict__ W2,
    const float* __restrict__ b2, const int* __restrict__ perm,
    const int* __restrict__ offsets, const int* __restrict__ numTiles,
    const int* __restrict__ tileExpert, const int* __restrict__ tileStart,
    float* __restrict__ out)
{
    __shared__ __align__(16) unsigned short Alds[BM * ASTRIDE];   // ~97 KB
    __shared__ float Ylds[BM][C_DIM];

    int b = blockIdx.x;
    if (b >= numTiles[0]) return;
    int e    = tileExpert[b];
    int tpos = tileStart[b];
    int rem  = offsets[e + 1] - tpos; if (rem > BM) rem = BM;

    int tid  = threadIdx.x;
    int lane = tid & 31, wave = tid >> 5;

    if (tid < BM * C_DIM) ((float*)Ylds)[tid] = 0.0f;

    // stage A tile (BM x 768) into LDS, f32 -> bf16, padded rows
    for (int idx = tid; idx < BM * (D_DIM / 8); idx += NTHREADS) {
        int row = idx / 96, ch = idx - row * 96;      // 96 chunks of 8 elems/row
        int tok = perm[tpos + (row < rem ? row : 0)];
        const float* src = emb + (size_t)tok * D_DIM + ch * 8;
        f32x4 f0 = *(const f32x4*)src;
        f32x4 f1 = *(const f32x4*)(src + 4);
        u32x4 pk;
        pk.x = f2bf(f0.x) | (f2bf(f0.y) << 16);
        pk.y = f2bf(f0.z) | (f2bf(f0.w) << 16);
        pk.z = f2bf(f1.x) | (f2bf(f1.y) << 16);
        pk.w = f2bf(f1.z) | (f2bf(f1.w) << 16);
        *(u32x4*)&Alds[row * ASTRIDE + ch * 8] = pk;
    }
    __syncthreads();

    int m_sub = wave & 3;        // which 16-token subtile (4 of them)
    int hgrp  = wave >> 2;       // 4 groups x 12 H-chunks
    int r  = lane & 15;
    int hi = lane >> 4;

    const unsigned short* Wexp = W1T + (size_t)e * H_DIM * D_DIM;
    const unsigned short* Arow = Alds + (m_sub * 16 + r) * ASTRIDE + hi * 8;

    float py0[8], py1[8];
    #pragma unroll
    for (int i = 0; i < 8; i++) { py0[i] = 0.f; py1[i] = 0.f; }

    for (int hp = 0; hp < 6; hp++) {
        int hb0 = (hgrp * 12 + hp * 2) * 16;
        int hb1 = hb0 + 16;
        v8f acc0 = {}; v8f acc1 = {};
        const unsigned short* B0 = Wexp + (size_t)(hb0 + r) * D_DIM + hi * 8;
        const unsigned short* B1 = Wexp + (size_t)(hb1 + r) * D_DIM + hi * 8;
        #pragma unroll 4
        for (int k = 0; k < NKCH; k++) {
            int kb = k * KCH;
            frag_u a;
            a.q[0] = *(const u32x4*)(Arow + kb);        // K {0..7 | 8..15}
            a.q[1] = *(const u32x4*)(Arow + kb + 16);   // K {16..23 | 24..31}
            frag_u fb0;
            fb0.q[0] = *(const u32x4*)(B0 + kb);        // K 0..7  (or 8..15)
            fb0.q[1] = *(const u32x4*)(B0 + kb + 16);   // K 16..23 (or 24..31)
            acc0 = __builtin_amdgcn_wmma_f32_16x16x32_bf16(
                       false, a.v, false, fb0.v, (short)0, acc0, false, false);
            frag_u fb1;
            fb1.q[0] = *(const u32x4*)(B1 + kb);
            fb1.q[1] = *(const u32x4*)(B1 + kb + 16);
            acc1 = __builtin_amdgcn_wmma_f32_16x16x32_bf16(
                       false, a.v, false, fb1.v, (short)0, acc1, false, false);
        }
        // fused bias + ReLU + second-layer (C=2) partial products
        float bias0 = b1[e * H_DIM + hb0 + r];
        float bias1 = b1[e * H_DIM + hb1 + r];
        f32x2 w20 = *(const f32x2*)&W2[((size_t)e * H_DIM + hb0 + r) * C_DIM];
        f32x2 w21 = *(const f32x2*)&W2[((size_t)e * H_DIM + hb1 + r) * C_DIM];
        #pragma unroll
        for (int i = 0; i < 8; i++) {
            float h0 = acc0[i] + bias0; h0 = h0 > 0.f ? h0 : 0.f;
            float h1 = acc1[i] + bias1; h1 = h1 > 0.f ? h1 : 0.f;
            py0[i] += h0 * w20.x + h1 * w21.x;
            py1[i] += h0 * w20.y + h1 * w21.y;
        }
    }

    // reduce over the 16 N-lanes (each half-wave independently)
    #pragma unroll
    for (int i = 0; i < 8; i++) {
        #pragma unroll
        for (int m = 1; m < 16; m <<= 1) {
            py0[i] += __shfl_xor(py0[i], m, 32);
            py1[i] += __shfl_xor(py1[i], m, 32);
        }
    }
    if (r == 0) {
        int mbase = m_sub * 16 + hi * 8;
        #pragma unroll
        for (int i = 0; i < 8; i++) {
            atomicAdd(&Ylds[mbase + i][0], py0[i]);
            atomicAdd(&Ylds[mbase + i][1], py1[i]);
        }
    }
    __syncthreads();

    if (tid < BM * C_DIM) {
        int m = tid >> 1, c = tid & 1;
        if (m < rem) {
            int tok = perm[tpos + m];
            out[(size_t)tok * C_DIM + c] = Ylds[m][c] + b2[e * C_DIM + c];
        }
    }
}

extern "C" void kernel_launch(void* const* d_in, const int* in_sizes, int n_in,
                              void* d_out, int out_size, void* d_ws, size_t ws_size,
                              hipStream_t stream) {
    const float* emb = (const float*)d_in[0];
    const int*   cid = (const int*)d_in[1];
    const float* W1  = (const float*)d_in[2];
    const float* b1  = (const float*)d_in[3];
    const float* W2  = (const float*)d_in[4];
    const float* b2  = (const float*)d_in[5];
    float* out = (float*)d_out;

    char* ws = (char*)d_ws;
    unsigned short* W1T = (unsigned short*)ws;                 // E*H*D bf16
    size_t off = (size_t)E_NUM * H_DIM * D_DIM * sizeof(unsigned short);
    int* counts     = (int*)(ws + off);
    int* cursors    = counts + E_NUM;
    int* offsets    = cursors + E_NUM;          // E+1
    int* numTiles   = offsets + (E_NUM + 1);
    int* tileExpert = numTiles + 1;             // MAX_TILES
    int* tileStart  = tileExpert + MAX_TILES;   // MAX_TILES
    int* perm       = tileStart + MAX_TILES;    // N ints

    hipMemsetAsync(counts, 0, 2 * E_NUM * sizeof(int), stream);

    k_w1_transpose<<<dim3(H_DIM / 32, D_DIM / 32, E_NUM), dim3(32, 8), 0, stream>>>(W1, W1T);
    k_hist<<<N_TOK / 256, 256, 0, stream>>>(cid, counts);
    k_scan<<<1, 32, 0, stream>>>(counts, offsets, numTiles, tileExpert, tileStart);
    k_scatter<<<N_TOK / 256, 256, 0, stream>>>(cid, offsets, cursors, perm);
    k_moe_gemm<<<MAX_TILES, NTHREADS, 0, stream>>>(emb, W1T, b1, W2, b2, perm,
                                                   offsets, numTiles, tileExpert, tileStart, out);
}
